// AttackHead_3015067041914
// MI455X (gfx1250) — matrix operand order repaired and checked by reading, not compiled, not run on hardware
//
#include <hip/hip_runtime.h>
#include <hip/hip_bf16.h>
#include <string.h>

typedef __attribute__((ext_vector_type(16))) __bf16 v16bf;
typedef __attribute__((ext_vector_type(8)))  __bf16 v8bf;
typedef __attribute__((ext_vector_type(8)))  float  v8f;

#define TILE_M 128   // rows (edges) per workgroup
#define D_EMB  256
#define K1     512   // 2*D
#define N1     256   // W1 cols
#define NG     128   // A1 cols
#define NA     64    // army actions (max_army_send)
#define BLOCK  128   // 4 waves; each wave owns 32 rows (2 M-fragments)

// ---------------------------------------------------------------------------
// Pack f32 weights (K x N row-major) into bf16, pre-swizzled into the WMMA
// B-fragment layout: for k-tile kt, n-tile nt, lane l holds 16 contiguous
// bf16: element e -> K = kt*32 + (e&7) + (e>=8 ? 16 : 0) + (l>=16 ? 8 : 0), N = nt*16 + (l&15).
// One 32-byte contiguous load per fragment at GEMM time.
// ---------------------------------------------------------------------------
__global__ void pack_bf16_kernel(const float* __restrict__ W,
                                 __bf16* __restrict__ out, int K, int N) {
  int idx = blockIdx.x * blockDim.x + threadIdx.x;
  if (idx >= K * N) return;
  int e    = idx & 15;
  int lane = (idx >> 4) & 31;
  int t    = idx >> 9;
  int NT   = N >> 4;
  int nt   = t % NT;
  int kt   = t / NT;
  int k = kt * 32 + (e & 7) + ((e & 8) ? 16 : 0) + ((lane & 16) ? 8 : 0);
  int n = nt * 16 + (lane & 15);
  out[idx] = (__bf16)W[k * N + n];
}

// A-fragment (16x32 bf16) from a row-major bf16 tile with row stride `stride`.
// Lane layout (ISA 7.12.2): lanes 0-15 row M=lane, halves = K 0..7 / 16..23;
// lanes 16-31 same rows, halves = K 8..15 / 24..31.
__device__ __forceinline__ v16bf load_a_frag(const __bf16* tile, int row,
                                             int kk, int stride, int lane) {
  int k0 = kk + ((lane & 16) ? 8 : 0);
  const __bf16* p = tile + row * stride + k0;
  v8bf lo = *(const v8bf*)(p);        // ds_load_b128
  v8bf hi = *(const v8bf*)(p + 16);   // ds_load_b128
  v16bf a;
#pragma unroll
  for (int i = 0; i < 8; ++i) { a[i] = lo[i]; a[i + 8] = hi[i]; }
  return a;
}

// B-fragment (32x16 bf16) from the pre-swizzled weight buffer: one aligned
// 32-byte contiguous load per lane (lowers to 2x global_load_b128).
__device__ __forceinline__ v16bf load_b_frag(const __bf16* __restrict__ packed,
                                             int kt, int nt, int NT, int lane) {
  const __bf16* p = packed + (((size_t)(kt * NT + nt) * 32 + lane) << 4);
  return *(const v16bf*)p;
}

#define WMMA_BF16(A, B, C) \
  __builtin_amdgcn_wmma_f32_16x16x32_bf16(false, (A), false, (B), (short)0, (C), false, false)

__global__ void __launch_bounds__(BLOCK)
attack_head_kernel(const float* __restrict__ emb,
                   const __bf16* __restrict__ W1p, const float* __restrict__ b1,
                   const float* __restrict__ W2,   const float* __restrict__ b2,
                   const __bf16* __restrict__ A1p, const float* __restrict__ a1,
                   const __bf16* __restrict__ A2p, const float* __restrict__ a2,
                   const int* __restrict__ edges,  const int* __restrict__ army,
                   float* __restrict__ edge_logits,
                   float* __restrict__ army_logits, int E) {
  __shared__ alignas(16) __bf16 Atile[TILE_M * K1];   // 128 KB: gathered edge_embed
  __shared__ alignas(16) __bf16 Gtile[TILE_M * NG];   //  32 KB: relu(A@A1+a1) bf16
  __shared__ int sSrc[TILE_M];
  __shared__ int sTgt[TILE_M];
  __shared__ int sMax[TILE_M];

  const int tid   = threadIdx.x;
  const int lane  = tid & 31;
  const int wid   = tid >> 5;
  const int tile0 = blockIdx.x * TILE_M;

  // ---- load edge endpoints + army caps for this tile --------------------
  if (tid < TILE_M) {
    int eIdx = tile0 + tid;
    if (eIdx >= E) eIdx = E - 1;        // clamp; writes are guarded below
    int2 se = ((const int2*)edges)[eIdx];   // one global_load_b64
    sSrc[tid] = se.x;
    sTgt[tid] = se.y;
    sMax[tid] = army[se.x] - 1;
  }
  __syncthreads();

  // ---- gather + f32->bf16 convert into LDS A-tile -----------------------
  // Each iteration: one global_load_b128 (4 floats) -> one ds_store_b64
  // (4 bf16 packed into a uint2).
  for (int i = tid; i < TILE_M * (K1 / 4); i += BLOCK) {
    int row  = i >> 7;                  // 128 four-float chunks per row
    int k    = (i & 127) << 2;
    int node = (k < D_EMB) ? sSrc[row] : sTgt[row];
    int koff = (k < D_EMB) ? k : (k - D_EMB);
    float4 v = *(const float4*)(emb + (size_t)node * D_EMB + koff);
    union { __bf16 h[4]; uint2 u; } pk;
    pk.h[0] = (__bf16)v.x; pk.h[1] = (__bf16)v.y;
    pk.h[2] = (__bf16)v.z; pk.h[3] = (__bf16)v.w;
    *(uint2*)(Atile + row * K1 + k) = pk.u;   // 8B-aligned ds_store_b64
  }
  __syncthreads();

  const int waveM   = wid * 32;            // each wave owns 32 rows
  const int halfSel = (lane & 16) ? 8 : 0; // C/D layout row offset
  const int nlane   = lane & 15;           // C/D layout column
  const int aRow0   = waveM + nlane;       // A-fragment 0 row for this lane
  const int aRow1   = waveM + 16 + nlane;  // A-fragment 1 row

  // ======================================================================
  // GEMM1: h = relu(A @ W1 + b1)   (128x512 @ 512x256), edge logit fused
  // 2x2 register blocking: 2 M-frags x 2 N-frags -> 4 WMMA per K-step,
  // 2 memory instructions per WMMA.
  // ======================================================================
  float part0[8], part1[8];
#pragma unroll
  for (int r = 0; r < 8; ++r) { part0[r] = 0.f; part1[r] = 0.f; }

  for (int nt = 0; nt < N1 / 16; nt += 2) {
    v8f c00 = {}, c01 = {}, c10 = {}, c11 = {};
#pragma unroll 2
    for (int kt = 0; kt < K1 / 32; ++kt) {
      v16bf a0 = load_a_frag(Atile, aRow0, kt * 32, K1, lane);
      v16bf a1 = load_a_frag(Atile, aRow1, kt * 32, K1, lane);
      v16bf b0 = load_b_frag(W1p, kt, nt,     N1 / 16, lane);
      v16bf b1 = load_b_frag(W1p, kt, nt + 1, N1 / 16, lane);
      c00 = WMMA_BF16(a0, b0, c00);
      c01 = WMMA_BF16(a0, b1, c01);
      c10 = WMMA_BF16(a1, b0, c10);
      c11 = WMMA_BF16(a1, b1, c11);
    }
    int   col0 = nt * 16 + nlane, col1 = col0 + 16;
    float bi0 = b1[col0], bi1 = b1[col1];
    float w20 = W2[col0], w21 = W2[col1];
#pragma unroll
    for (int r = 0; r < 8; ++r) {
      part0[r] += fmaxf(c00[r] + bi0, 0.f) * w20 + fmaxf(c01[r] + bi1, 0.f) * w21;
      part1[r] += fmaxf(c10[r] + bi0, 0.f) * w20 + fmaxf(c11[r] + bi1, 0.f) * w21;
    }
  }
  // reduce the 16 column-lanes of each half-wave
#pragma unroll
  for (int r = 0; r < 8; ++r) {
#pragma unroll
    for (int off = 8; off >= 1; off >>= 1) {
      part0[r] += __shfl_xor(part0[r], off, 32);
      part1[r] += __shfl_xor(part1[r], off, 32);
    }
  }
  if (nlane == 0) {
    float bb = b2[0];
#pragma unroll
    for (int r = 0; r < 8; ++r) {
      int r0 = tile0 + waveM + halfSel + r;
      int r1 = r0 + 16;
      if (r0 < E) edge_logits[r0] = part0[r] + bb;
      if (r1 < E) edge_logits[r1] = part1[r] + bb;
    }
  }

  // ======================================================================
  // GEMM-A1: g = relu(A @ A1 + a1)  (128x512 @ 512x128) -> bf16 LDS
  // ======================================================================
  for (int nt = 0; nt < NG / 16; nt += 2) {
    v8f c00 = {}, c01 = {}, c10 = {}, c11 = {};
#pragma unroll 2
    for (int kt = 0; kt < K1 / 32; ++kt) {
      v16bf a0 = load_a_frag(Atile, aRow0, kt * 32, K1, lane);
      v16bf a1 = load_a_frag(Atile, aRow1, kt * 32, K1, lane);
      v16bf b0 = load_b_frag(A1p, kt, nt,     NG / 16, lane);
      v16bf b1 = load_b_frag(A1p, kt, nt + 1, NG / 16, lane);
      c00 = WMMA_BF16(a0, b0, c00);
      c01 = WMMA_BF16(a0, b1, c01);
      c10 = WMMA_BF16(a1, b0, c10);
      c11 = WMMA_BF16(a1, b1, c11);
    }
    int   col0 = nt * 16 + nlane, col1 = col0 + 16;
    float bi0 = a1[col0], bi1 = a1[col1];
#pragma unroll
    for (int r = 0; r < 8; ++r) {
      int row0 = waveM + halfSel + r;
      int row1 = row0 + 16;
      Gtile[row0 * NG + col0] = (__bf16)fmaxf(c00[r] + bi0, 0.f);
      Gtile[row0 * NG + col1] = (__bf16)fmaxf(c01[r] + bi1, 0.f);
      Gtile[row1 * NG + col0] = (__bf16)fmaxf(c10[r] + bi0, 0.f);
      Gtile[row1 * NG + col1] = (__bf16)fmaxf(c11[r] + bi1, 0.f);
    }
  }
  __syncthreads();

  // ======================================================================
  // GEMM-A2: army = g @ A2 + a2  (128x128 @ 128x64), mask + store
  // ======================================================================
  for (int nt = 0; nt < NA / 16; nt += 2) {
    v8f c00 = {}, c01 = {}, c10 = {}, c11 = {};
#pragma unroll
    for (int kt = 0; kt < NG / 32; ++kt) {
      v16bf a0 = load_a_frag(Gtile, aRow0, kt * 32, NG, lane);
      v16bf a1 = load_a_frag(Gtile, aRow1, kt * 32, NG, lane);
      v16bf b0 = load_b_frag(A2p, kt, nt,     NA / 16, lane);
      v16bf b1 = load_b_frag(A2p, kt, nt + 1, NA / 16, lane);
      c00 = WMMA_BF16(a0, b0, c00);
      c01 = WMMA_BF16(a0, b1, c01);
      c10 = WMMA_BF16(a1, b0, c10);
      c11 = WMMA_BF16(a1, b1, c11);
    }
    int   col0 = nt * 16 + nlane, col1 = col0 + 16;
    float bi0 = a2[col0], bi1 = a2[col1];
#pragma unroll
    for (int r = 0; r < 8; ++r) {
      int rl0 = waveM + halfSel + r;
      int rl1 = rl0 + 16;
      int rg0 = tile0 + rl0;
      int rg1 = tile0 + rl1;
      float v00 = c00[r] + bi0, v01 = c01[r] + bi1;
      float v10 = c10[r] + bi0, v11 = c11[r] + bi1;
      int m0 = sMax[rl0], m1 = sMax[rl1];
      if (col0 >= m0) v00 = -1000000000.0f;
      if (col1 >= m0) v01 = -1000000000.0f;
      if (col0 >= m1) v10 = -1000000000.0f;
      if (col1 >= m1) v11 = -1000000000.0f;
      if (rg0 < E) {
        army_logits[(size_t)rg0 * NA + col0] = v00;
        army_logits[(size_t)rg0 * NA + col1] = v01;
      }
      if (rg1 < E) {
        army_logits[(size_t)rg1 * NA + col0] = v10;
        army_logits[(size_t)rg1 * NA + col1] = v11;
      }
    }
  }
}

extern "C" void kernel_launch(void* const* d_in, const int* in_sizes, int n_in,
                              void* d_out, int out_size, void* d_ws, size_t ws_size,
                              hipStream_t stream) {
  const float* emb = (const float*)d_in[0];
  const float* W1  = (const float*)d_in[1];
  const float* b1  = (const float*)d_in[2];
  const float* W2  = (const float*)d_in[3];
  const float* b2  = (const float*)d_in[4];
  const float* A1  = (const float*)d_in[5];
  const float* a1  = (const float*)d_in[6];
  const float* A2  = (const float*)d_in[7];
  const float* a2  = (const float*)d_in[8];
  const int* edges = (const int*)d_in[9];
  const int* army  = (const int*)d_in[10];
  const int  E     = in_sizes[9] / 2;

  // workspace: packed bf16 weights (re-packed every launch; deterministic)
  char* ws = (char*)d_ws;
  __bf16* W1p = (__bf16*)(ws);
  __bf16* A1p = (__bf16*)(ws + (size_t)K1 * N1 * 2);
  __bf16* A2p = (__bf16*)(ws + (size_t)K1 * N1 * 2 + (size_t)K1 * NG * 2);

  const int T = 256;
  pack_bf16_kernel<<<(K1 * N1 + T - 1) / T, T, 0, stream>>>(W1, W1p, K1, N1);
  pack_bf16_kernel<<<(K1 * NG + T - 1) / T, T, 0, stream>>>(A1, A1p, K1, NG);
  pack_bf16_kernel<<<(NG * NA + T - 1) / T, T, 0, stream>>>(A2, A2p, NG, NA);

  float* edge_logits = (float*)d_out;
  float* army_logits = (float*)d_out + E;
  int grid = (E + TILE_M - 1) / TILE_M;
  attack_head_kernel<<<grid, BLOCK, 0, stream>>>(
      emb, W1p, b1, W2, b2, A1p, a1, A2p, a2, edges, army,
      edge_logits, army_logits, E);
}
